// FCGN_29721173689131
// MI455X (gfx1250) — matrix operand order repaired and checked by reading, not compiled, not run on hardware
//
#include <hip/hip_runtime.h>
#include <hip/hip_bf16.h>

typedef __attribute__((ext_vector_type(16))) _Float16 v16h;
typedef __attribute__((ext_vector_type(8)))  float    v8f;

#define NHID 256
#define NNODE 16
#define NIN 14

union AFrag { v16h v; unsigned u[8]; };
union BFrag { v16h v; float4 f4[2]; };

// A-fragment (16x32 f16, row-major source, stride lda) per ISA 16-bit A layout:
// lane m=lane&15 is row; hi=lane>>4 selects K-halves; VGPR p holds K pair
// kb = ((p>>2)<<4) + hi*8 + ((p&3)<<1).
__device__ __forceinline__ v16h load_a_frag(const _Float16* A, int lda, int k0, int lane) {
    AFrag af;
    int m = lane & 15, hi = lane >> 4;
    const unsigned* row = (const unsigned*)(A + m * lda + k0);
#pragma unroll
    for (int p = 0; p < 8; ++p) {
        int kb = ((p >> 2) << 4) + hi * 8 + ((p & 3) << 1);
        af.u[p] = row[kb >> 1];
    }
    return af.v;
}

// B-fragment (32x16 f16) from row-major K x 256 weights: lane n=lane&15 is the
// column; element s holds K = hi*8+s (s<8) / 16+hi*8+(s-8) (s>=8), matching A.
__device__ __forceinline__ v16h load_b_frag(const _Float16* B, int k0, int n0, int lane) {
    v16h bv;
    int nn = lane & 15, hi = lane >> 4;
    const _Float16* col = B + k0 * NHID + n0 + nn;
#pragma unroll
    for (int s = 0; s < 16; ++s) {
        int k = (s < 8) ? (hi * 8 + s) : (16 + hi * 8 + (s - 8));
        bv[s] = col[k * NHID];
    }
    return bv;
}

// C = A(16xKDIM, f16 in LDS) * B(KDIM x 256, f16 row-major) + bias, opt relu.
// 8 waves x 2 N-tiles; f32 accumulate. All policy flags are compile-time to
// keep the epilogue branch-free.
template<int KDIM, bool BIAS, bool RELU, bool O32, bool O16>
__device__ __forceinline__ void gemm16(const _Float16* __restrict__ A,
                                       const _Float16* __restrict__ B,
                                       const float* __restrict__ bias,
                                       float* __restrict__ outF32,
                                       _Float16* __restrict__ outF16, int tid) {
    int lane = tid & 31, w = tid >> 5;
    int n0a = w * 32, n0b = w * 32 + 16;
    v8f acc0 = {}; v8f acc1 = {};
#pragma unroll
    for (int k0 = 0; k0 < KDIM; k0 += 32) {
        v16h a  = load_a_frag(A, KDIM, k0, lane);
        v16h b0 = load_b_frag(B, k0, n0a, lane);
        v16h b1 = load_b_frag(B, k0, n0b, lane);
        acc0 = __builtin_amdgcn_wmma_f32_16x16x32_f16(false, a, false, b0, (short)0, acc0, false, false);
        acc1 = __builtin_amdgcn_wmma_f32_16x16x32_f16(false, a, false, b1, (short)0, acc1, false, false);
    }
    int nn = lane & 15, hi = lane >> 4;
    float bb0 = BIAS ? bias[n0a + nn] : 0.f;
    float bb1 = BIAS ? bias[n0b + nn] : 0.f;
#pragma unroll
    for (int e = 0; e < 8; ++e) {
        int m = e + 8 * hi;   // C layout: VGPR e -> M = e (lanes 0-15) / e+8 (lanes 16-31)
        float v0 = acc0[e] + bb0;
        float v1 = acc1[e] + bb1;
        if (RELU) { v0 = fmaxf(v0, 0.f); v1 = fmaxf(v1, 0.f); }
        if (O32) { outF32[m * NHID + n0a + nn] = v0; outF32[m * NHID + n0b + nn] = v1; }
        if (O16) { outF16[m * NHID + n0a + nn] = (_Float16)v0; outF16[m * NHID + n0b + nn] = (_Float16)v1; }
    }
}

// Fused p/q GEMM: both read the same A operand (h in f16), so load the A
// fragment once per k-step and feed 4 back-to-back WMMAs.
__device__ __forceinline__ void gemm_pq(const _Float16* __restrict__ A,
                                        const _Float16* __restrict__ Wp,
                                        const _Float16* __restrict__ Wq,
                                        float* __restrict__ sP,
                                        float* __restrict__ sQ, int tid) {
    int lane = tid & 31, w = tid >> 5;
    int n0a = w * 32, n0b = w * 32 + 16;
    v8f ap0 = {}; v8f ap1 = {}; v8f aq0 = {}; v8f aq1 = {};
#pragma unroll
    for (int k0 = 0; k0 < NHID; k0 += 32) {
        v16h a   = load_a_frag(A, NHID, k0, lane);
        v16h bp0 = load_b_frag(Wp, k0, n0a, lane);
        v16h bp1 = load_b_frag(Wp, k0, n0b, lane);
        v16h bq0 = load_b_frag(Wq, k0, n0a, lane);
        v16h bq1 = load_b_frag(Wq, k0, n0b, lane);
        ap0 = __builtin_amdgcn_wmma_f32_16x16x32_f16(false, a, false, bp0, (short)0, ap0, false, false);
        ap1 = __builtin_amdgcn_wmma_f32_16x16x32_f16(false, a, false, bp1, (short)0, ap1, false, false);
        aq0 = __builtin_amdgcn_wmma_f32_16x16x32_f16(false, a, false, bq0, (short)0, aq0, false, false);
        aq1 = __builtin_amdgcn_wmma_f32_16x16x32_f16(false, a, false, bq1, (short)0, aq1, false, false);
    }
    int nn = lane & 15, hi = lane >> 4;
#pragma unroll
    for (int e = 0; e < 8; ++e) {
        int m = e + 8 * hi;
        sP[m * NHID + n0a + nn] = ap0[e];
        sP[m * NHID + n0b + nn] = ap1[e];
        sQ[m * NHID + n0a + nn] = aq0[e];
        sQ[m * NHID + n0b + nn] = aq1[e];
    }
}

// Fused edge GEMM for 4 receivers i0..i0+3: all share B (= LDS-resident Wm2),
// so each k-step loads one B-fragment pair and 4 A-fragments and issues 8
// back-to-back WMMAs. Epilogue: bias + relu + masked (j != i) reduction over
// M per receiver, writing edges rows i0..i0+3.
__device__ __forceinline__ void gemm_edge4(const _Float16* __restrict__ Ag,
                                           const _Float16* __restrict__ B,
                                           const float* __restrict__ bm2,
                                           int i0, float* __restrict__ sE, int tid) {
    int lane = tid & 31, w = tid >> 5;
    int n0a = w * 32, n0b = w * 32 + 16;
    v8f acc[4][2] = {};
#pragma unroll
    for (int k0 = 0; k0 < NHID; k0 += 32) {
        v16h b0 = load_b_frag(B, k0, n0a, lane);
        v16h b1 = load_b_frag(B, k0, n0b, lane);
#pragma unroll
        for (int r = 0; r < 4; ++r) {
            v16h a = load_a_frag(Ag + r * (NNODE * NHID), NHID, k0, lane);
            acc[r][0] = __builtin_amdgcn_wmma_f32_16x16x32_f16(false, a, false, b0, (short)0, acc[r][0], false, false);
            acc[r][1] = __builtin_amdgcn_wmma_f32_16x16x32_f16(false, a, false, b1, (short)0, acc[r][1], false, false);
        }
    }
    int nn = lane & 15, hi = lane >> 4;
    float bb0 = bm2[n0a + nn], bb1 = bm2[n0b + nn];
#pragma unroll
    for (int r = 0; r < 4; ++r) {
        float p0 = 0.f, p1 = 0.f;
#pragma unroll
        for (int e = 0; e < 8; ++e) {
            int j = e + 8 * hi;
            float v0 = fmaxf(acc[r][0][e] + bb0, 0.f);
            float v1 = fmaxf(acc[r][1][e] + bb1, 0.f);
            if (j != i0 + r) { p0 += v0; p1 += v1; }
        }
        p0 += __shfl_xor(p0, 16);
        p1 += __shfl_xor(p1, 16);
        if (lane < 16) {
            sE[(i0 + r) * NHID + n0a + nn] = p0;
            sE[(i0 + r) * NHID + n0b + nn] = p1;
        }
    }
}

__global__ void f32_to_f16_kernel(const float* __restrict__ src, _Float16* __restrict__ dst, int n) {
    int t = blockIdx.x * blockDim.x + threadIdx.x;
    if (t < n) dst[t] = (_Float16)src[t];
}

// We (14x256) -> zero-padded 32x256 f16 so the encoder uses the K=32 WMMA path.
__global__ void pad_we_kernel(const float* __restrict__ src, _Float16* __restrict__ dst) {
    int t = blockIdx.x * blockDim.x + threadIdx.x;
    if (t < 32 * 256) {
        int r = t >> 8, c = t & 255;
        dst[t] = (r < NIN) ? (_Float16)src[r * 256 + c] : (_Float16)0.f;
    }
}

__global__ __launch_bounds__(256) void fcgn_kernel(
    const float* __restrict__ towers, const float* __restrict__ be,
    const float* __restrict__ bm1, const float* __restrict__ bm2,
    const float* __restrict__ bu1, const float* __restrict__ bu2,
    const float* __restrict__ bo1, const float* __restrict__ Wo2,
    const float* __restrict__ bo2, const int* __restrict__ kptr,
    const _Float16* __restrict__ WeP, const _Float16* __restrict__ Wm1h,
    const _Float16* __restrict__ Wm2h, const _Float16* __restrict__ Wu1h,
    const _Float16* __restrict__ Wu2h, const _Float16* __restrict__ Wo1h,
    float* __restrict__ out) {
    extern __shared__ char smem[];
    float*    sH   = (float*)smem;            // 16x256 f32 node features (master)
    float*    sE   = sH + 4096;               // 16x256 f32 edges / scratch
    float*    sP   = sE + 4096;               // 16x256 f32 sender part (+logit scratch)
    float*    sQ   = sP + 4096;               // 16x256 f32 receiver part
    float*    sB1  = sQ + 4096;               // bm1 staged
    float*    sB2  = sB1 + 256;               // bm2 staged
    _Float16* sWm2 = (_Float16*)(sB2 + 256);  // 256x256 f16 edge weight (hot)
    _Float16* sXg  = sWm2 + 65536;            // 4 x (16x256) f16 A_i group staging
    _Float16* sY   = sXg + 4 * 4096;          // 16x256 f16 A-operand staging
    _Float16* sHh  = sY + 4096;               // 16x256 f16 copy of h

    int tid = threadIdx.x;
    int n = blockIdx.x;
    int kiter = *kptr;

    // Stage hot edge weight Wm2 into LDS (used 16x per iteration), plus biases.
    {
        const float4* src = (const float4*)Wm2h;
        float4* dst = (float4*)sWm2;
        for (int t = tid; t < 8192; t += 256) dst[t] = src[t];
        sB1[tid] = bm1[tid];
        sB2[tid] = bm2[tid];
    }
    // Encoder operand: towers row-padded 16x32 f16 (staged in sXg).
    for (int t = tid; t < 16 * 32; t += 256) {
        int m = t >> 5, c = t & 31;
        sXg[t] = (c < NIN) ? (_Float16)towers[n * (NNODE * NIN) + m * NIN + c] : (_Float16)0.f;
    }
    __syncthreads();
    gemm16<32, true, false, true, true>(sXg, WeP, be, sH, sHh, tid); // h = towers@We + be
    __syncthreads();

    for (int it = 0; it < kiter; ++it) {
        gemm_pq(sHh, Wm1h, Wm1h + 256 * NHID, sP, sQ, tid);  // p/q = h @ Wm1[:nh]/[nh:]
        __syncthreads();
        for (int i0 = 0; i0 < NNODE; i0 += 4) {
            // Build 4 A_i tiles: A[r][j,c] = relu(p[i0+r,c] + q[j,c] + bm1[c])
            for (int t = tid; t < 4 * 4096; t += 256) {
                int r = t >> 12, j = (t >> 8) & 15, c = t & 255;
                float pre = sP[(i0 + r) * NHID + c] + sQ[j * NHID + c] + sB1[c];
                sXg[t] = (_Float16)fmaxf(pre, 0.f);
            }
            __syncthreads();
            gemm_edge4(sXg, sWm2, sB2, i0, sE, tid);  // edges[i0..i0+3]
            __syncthreads();
        }
        for (int t = tid; t < 4096; t += 256)         // x = h + edges
            sXg[t] = (_Float16)(sH[t] + sE[t]);
        __syncthreads();
        gemm16<256, true, true,  false, true>(sXg, Wu1h, bu1, nullptr, sY, tid); // t = relu(x@Wu1+bu1)
        __syncthreads();
        gemm16<256, true, false, true,  true>(sY, Wu2h, bu2, sH, sHh, tid);      // h = t@Wu2+bu2
        __syncthreads();
    }

    // Output head: g = relu(h@Wo1+bo1); logit_i = g_i . Wo2 + bo2
    gemm16<256, true, true, true, false>(sHh, Wo1h, bo1, sE, nullptr, tid);
    __syncthreads();
    int lane = tid & 31, w = tid >> 5;
#pragma unroll
    for (int rs = 0; rs < 2; ++rs) {
        int r = w + 8 * rs;
        float s = 0.f;
#pragma unroll
        for (int c = 0; c < NHID; c += 32) s += sE[r * NHID + c + lane] * Wo2[c + lane];
#pragma unroll
        for (int off = 16; off > 0; off >>= 1) s += __shfl_xor(s, off);
        if (lane == 0) sP[r] = s + bo2[0];
    }
    __syncthreads();
    if (tid == 0) {
        float prod = 1.f;
#pragma unroll
        for (int r = 0; r < NNODE; ++r) prod *= 1.f / (1.f + __expf(-sP[r]));
        out[n] = prod;
    }
}

extern "C" void kernel_launch(void* const* d_in, const int* in_sizes, int n_in,
                              void* d_out, int out_size, void* d_ws, size_t ws_size,
                              hipStream_t stream) {
    const float* towers = (const float*)d_in[0];
    const float* We  = (const float*)d_in[1];
    const float* be  = (const float*)d_in[2];
    const float* Wm1 = (const float*)d_in[3];
    const float* bm1 = (const float*)d_in[4];
    const float* Wm2 = (const float*)d_in[5];
    const float* bm2 = (const float*)d_in[6];
    const float* Wu1 = (const float*)d_in[7];
    const float* bu1 = (const float*)d_in[8];
    const float* Wu2 = (const float*)d_in[9];
    const float* bu2 = (const float*)d_in[10];
    const float* Wo1 = (const float*)d_in[11];
    const float* bo1 = (const float*)d_in[12];
    const float* Wo2 = (const float*)d_in[13];
    const float* bo2 = (const float*)d_in[14];
    const int*   kpt = (const int*)d_in[15];
    float* out = (float*)d_out;

    _Float16* ws   = (_Float16*)d_ws;
    _Float16* WeP  = ws;                 // 32x256
    _Float16* Wm1h = WeP  + 8192;        // 512x256
    _Float16* Wm2h = Wm1h + 131072;      // 256x256
    _Float16* Wu1h = Wm2h + 65536;
    _Float16* Wu2h = Wu1h + 65536;
    _Float16* Wo1h = Wu2h + 65536;

    pad_we_kernel    <<<32,  256, 0, stream>>>(We,  WeP);
    f32_to_f16_kernel<<<512, 256, 0, stream>>>(Wm1, Wm1h, 131072);
    f32_to_f16_kernel<<<256, 256, 0, stream>>>(Wm2, Wm2h, 65536);
    f32_to_f16_kernel<<<256, 256, 0, stream>>>(Wu1, Wu1h, 65536);
    f32_to_f16_kernel<<<256, 256, 0, stream>>>(Wu2, Wu2h, 65536);
    f32_to_f16_kernel<<<256, 256, 0, stream>>>(Wo1, Wo1h, 65536);

    size_t smem = (4 * 4096 + 512) * sizeof(float) + (65536 + 6 * 4096) * sizeof(_Float16);
    fcgn_kernel<<<512, 256, smem, stream>>>(towers, be, bm1, bm2, bu1, bu2, bo1,
                                            Wo2, bo2, kpt, WeP, Wm1h, Wm2h, Wu1h,
                                            Wu2h, Wo1h, out);
}
